// Policy_56014963475200
// MI455X (gfx1250) — matrix operand order repaired and checked by reading, not compiled, not run on hardware
//
#include <hip/hip_runtime.h>
#include <hip/hip_bf16.h>

typedef __attribute__((ext_vector_type(16))) _Float16 v16h;
typedef __attribute__((ext_vector_type(8)))  _Float16 v8h;
typedef __attribute__((ext_vector_type(8)))  float    v8f;

#define HID 128
#define S1 50
#define S2 10

// ---------------------------------------------------------------------------
// Fragment loaders (wave32, v_wmma_f32_16x16x32_f16 operand layouts)
// A matrix 16x32 f16, row-major source with leading dim ldk (elements):
//   lanes 0-15 : row=lane,    K = 0..15
//   lanes 16-31: row=lane-16, K = 16..31
__device__ __forceinline__ v16h load_frag_A(const _Float16* base, int ldk, int lane) {
    int row = lane & 15;
    int kb  = (lane >> 4) * 16;
    const _Float16* p = base + row * ldk + kb;
    v8h lo = *(const v8h*)(p);
    v8h hi = *(const v8h*)(p + 8);
    v16h r;
#pragma unroll
    for (int i = 0; i < 8; ++i) { r[i] = lo[i]; r[i + 8] = hi[i]; }
    return r;
}

// B matrix 32x16 f16 supplied as B^T row-major [N][K] (i.e. the weight matrix
// itself, since every layer computes x @ W^T):
//   VGPR r: lanes 0-15 hold K=2r,2r+1 ; lanes 16-31 hold K=16+2r,16+2r+1
//   column N = lane & 15  -> lane reads 16 contiguous K values (32 bytes)
__device__ __forceinline__ v16h load_frag_B(const _Float16* baseT, int ldk, int lane) {
    int col = lane & 15;
    int kb  = (lane >> 4) * 16;
    const _Float16* p = baseT + col * ldk + kb;
    v8h lo = *(const v8h*)(p);
    v8h hi = *(const v8h*)(p + 8);
    v16h r;
#pragma unroll
    for (int i = 0; i < 8; ++i) { r[i] = lo[i]; r[i + 8] = hi[i]; }
    return r;
}

// ---------------------------------------------------------------------------
// Kernel 0: convert weights to f16 in workspace; fold constant column into bias.
__global__ void convert_kernel(const float* __restrict__ W2,
                               const float* __restrict__ w_a,
                               const float* __restrict__ b_a,
                               const float* __restrict__ w_b,
                               const int*   __restrict__ ps,
                               const int*   __restrict__ ne,
                               _Float16* __restrict__ W2h,
                               _Float16* __restrict__ WAh,
                               _Float16* __restrict__ WBh,
                               float* __restrict__ biasA) {
    int t      = blockIdx.x * blockDim.x + threadIdx.x;
    int stride = gridDim.x * blockDim.x;
    for (int i = t; i < HID * HID; i += stride) W2h[i] = (_Float16)W2[i];
    for (int i = t; i < HID * HID; i += stride) {
        int j = i >> 7, k = i & 127;
        WAh[i] = (_Float16)w_a[j * 129 + k];
    }
    for (int i = t; i < 64 * HID; i += stride) WBh[i] = (_Float16)w_b[i];
    float rs = (float)(ps[0] - ne[0]);
    for (int j = t; j < HID; j += stride)
        biasA[j] = b_a[j] + rs * w_a[j * 129 + HID];
}

// ---------------------------------------------------------------------------
// Kernel 1: enc1 — one wave per node. Gather 50 neighbor feature rows, mean
// with self (gcn=True => /51), 3->128 matvec + ReLU into f32 h1.
__global__ void __launch_bounds__(256) enc1_kernel(const float* __restrict__ features,
                                                   const int*   __restrict__ neigh1,
                                                   const float* __restrict__ W1,
                                                   float* __restrict__ h1, int N) {
    int wave = threadIdx.x >> 5;
    int lane = threadIdx.x & 31;
    int n = blockIdx.x * 8 + wave;
    if (n >= N) return;

    float s0 = 0.f, s1 = 0.f, s2 = 0.f;
    for (int s = lane; s < S1; s += 32) {
        int idx = neigh1[n * S1 + s];
        const float* f = features + (size_t)idx * 3;
        s0 += f[0]; s1 += f[1]; s2 += f[2];
    }
#pragma unroll
    for (int off = 16; off > 0; off >>= 1) {
        s0 += __shfl_xor(s0, off, 32);
        s1 += __shfl_xor(s1, off, 32);
        s2 += __shfl_xor(s2, off, 32);
    }
    const float* fs = features + (size_t)n * 3;
    const float inv = 1.0f / (float)(S1 + 1);
    float a0 = (s0 + fs[0]) * inv;
    float a1 = (s1 + fs[1]) * inv;
    float a2 = (s2 + fs[2]) * inv;

#pragma unroll
    for (int q = 0; q < 4; ++q) {
        int j = lane + q * 32;
        float v = a0 * W1[j * 3 + 0] + a1 * W1[j * 3 + 1] + a2 * W1[j * 3 + 2];
        h1[(size_t)n * HID + j] = fmaxf(v, 0.f);   // v_max_num_f32
    }
}

// ---------------------------------------------------------------------------
// Kernel 2: enc2 + MLP fused. One wave per 16-node tile; 4 waves / block.
// Stages: gather+mean -> LDS f16 A tile -> WMMA (W2) -> WMMA (w_a + biasA)
// -> WMMA (w_b + b_b) -> per-node 64-dot with w_c -> scores.
__global__ void __launch_bounds__(128) enc2_mlp_kernel(const int*   __restrict__ nodes,
                                                       const float* __restrict__ h1,
                                                       const int*   __restrict__ neigh2,
                                                       const _Float16* __restrict__ W2h,
                                                       const _Float16* __restrict__ WAh,
                                                       const _Float16* __restrict__ WBh,
                                                       const float* __restrict__ biasA,
                                                       const float* __restrict__ b_b,
                                                       const float* __restrict__ w_c,
                                                       const float* __restrict__ b_c,
                                                       float* __restrict__ scores, int N) {
    __shared__ _Float16 smem[4 * 2 * 16 * HID];  // 32 KB: 2 ping-pong tiles per wave
    int wave = threadIdx.x >> 5;
    int lane = threadIdx.x & 31;
    int tile = blockIdx.x * 4 + wave;
    if (tile * 16 >= N) return;            // wave-uniform: EXEC stays all-ones

    _Float16* A  = smem + wave * (2 * 16 * HID);
    _Float16* Bf = A + 16 * HID;

    // ---- stage 0: agg2 = (sum_{s<10} h1[neigh2[node]] + h1[node]) / 11 -> A (f16)
    int col4 = lane * 4;                   // each lane owns 4 of the 128 columns
    const float inv11 = 1.0f / (float)(S2 + 1);
    for (int i = 0; i < 16; ++i) {
        int q = tile * 16 + i;
        if (q >= N) q = N - 1;
        int n = nodes[q];
        float4 acc = *(const float4*)(h1 + (size_t)n * HID + col4);
        for (int s = 0; s < S2; ++s) {
            int idx = neigh2[n * S2 + s];
            float4 v = *(const float4*)(h1 + (size_t)idx * HID + col4);
            acc.x += v.x; acc.y += v.y; acc.z += v.z; acc.w += v.w;
        }
        _Float16* dst = A + i * HID + col4;
        dst[0] = (_Float16)(acc.x * inv11);
        dst[1] = (_Float16)(acc.y * inv11);
        dst[2] = (_Float16)(acc.z * inv11);
        dst[3] = (_Float16)(acc.w * inv11);
    }
    __builtin_amdgcn_wave_barrier();

    // ---- generic WMMA layer: dst = relu(Asrc[16xK=128] @ Bw^T + bias)
    auto gemm_layer = [&](const _Float16* Asrc, const _Float16* Bw,
                          const float* bias, _Float16* dst, int njt, int dstride) {
        int nn = lane & 15;
        int mo = (lane >> 4) * 8;
        for (int jt = 0; jt < njt; ++jt) {
            v8f c = {};
#pragma unroll
            for (int kc = 0; kc < 4; ++kc) {
                v16h a = load_frag_A(Asrc + kc * 32, HID, lane);
                v16h b = load_frag_B(Bw + (size_t)jt * 16 * HID + kc * 32, HID, lane);
                c = __builtin_amdgcn_wmma_f32_16x16x32_f16(
                        false, a, false, b, (short)0, c, false, false);
            }
            float bv = bias ? bias[jt * 16 + nn] : 0.0f;
#pragma unroll
            for (int r = 0; r < 8; ++r) {
                // bias==nullptr folds at inline time -> single v_max_num_f32
                float v = bias ? (c[r] + bv) : c[r];
                v = fmaxf(v, 0.f);
                dst[(mo + r) * dstride + jt * 16 + nn] = (_Float16)v;
            }
        }
        __builtin_amdgcn_wave_barrier();
    };

    gemm_layer(A,  W2h, nullptr, Bf, 8, HID);   // emb  = relu(agg2 @ W2^T)       -> Bf
    gemm_layer(Bf, WAh, biasA,   A,  8, HID);   // xa   = relu(emb @ Wa^T + b')   -> A
    gemm_layer(A,  WBh, b_b,     Bf, 4, 64);    // xb   = relu(xa @ Wb^T + b_b)   -> Bf [16][64]

    // ---- stage 4: score = xb . w_c + b_c
    if (lane < 16) {
        int q = tile * 16 + lane;
        if (q < N) {
            float s = b_c[0];
            const _Float16* xb = Bf + lane * 64;
#pragma unroll 8
            for (int k = 0; k < 64; ++k) s += (float)xb[k] * w_c[k];
            scores[q] = s;
        }
    }
}

// ---------------------------------------------------------------------------
// Kernel 3: softmax over all N scores (single block, 3 strided passes).
__global__ void __launch_bounds__(1024) softmax_kernel(const float* __restrict__ scores,
                                                       float* __restrict__ out, int N) {
    __shared__ float red[1024];
    int t = threadIdx.x;

    float m = -3.0e38f;
    for (int i = t; i < N; i += 1024) m = fmaxf(m, scores[i]);
    red[t] = m;
    __syncthreads();
    for (int off = 512; off > 0; off >>= 1) {
        if (t < off) red[t] = fmaxf(red[t], red[t + off]);
        __syncthreads();
    }
    float gmax = red[0];
    __syncthreads();

    float s = 0.f;
    for (int i = t; i < N; i += 1024) s += __expf(scores[i] - gmax);
    red[t] = s;
    __syncthreads();
    for (int off = 512; off > 0; off >>= 1) {
        if (t < off) red[t] += red[t + off];
        __syncthreads();
    }
    float inv = 1.0f / red[0];

    for (int i = t; i < N; i += 1024) out[i] = __expf(scores[i] - gmax) * inv;
}

// ---------------------------------------------------------------------------
extern "C" void kernel_launch(void* const* d_in, const int* in_sizes, int n_in,
                              void* d_out, int out_size, void* d_ws, size_t ws_size,
                              hipStream_t stream) {
    const int*   nodes    = (const int*)d_in[0];
    const float* features = (const float*)d_in[1];
    const int*   neigh1   = (const int*)d_in[2];
    const int*   neigh2   = (const int*)d_in[3];
    const float* W1       = (const float*)d_in[4];
    const float* W2       = (const float*)d_in[5];
    const float* w_a      = (const float*)d_in[6];
    const float* b_a      = (const float*)d_in[7];
    const float* w_b      = (const float*)d_in[8];
    const float* b_b      = (const float*)d_in[9];
    const float* w_c      = (const float*)d_in[10];
    const float* b_c      = (const float*)d_in[11];
    const int*   ps       = (const int*)d_in[12];
    const int*   ne       = (const int*)d_in[13];

    const int N = in_sizes[0];   // 100000 query nodes (= all nodes)

    // ---- carve workspace (256B-aligned slices)
    size_t off = 0;
    auto carve = [&](size_t bytes) -> char* {
        char* p = (char*)d_ws + off;
        off += (bytes + 255) & ~(size_t)255;
        return p;
    };
    float*    h1     = (float*)   carve((size_t)N * HID * sizeof(float));
    float*    scores = (float*)   carve((size_t)N * sizeof(float));
    _Float16* W2h    = (_Float16*)carve((size_t)HID * HID * sizeof(_Float16));
    _Float16* WAh    = (_Float16*)carve((size_t)HID * HID * sizeof(_Float16));
    _Float16* WBh    = (_Float16*)carve((size_t)64 * HID * sizeof(_Float16));
    float*    biasA  = (float*)   carve((size_t)HID * sizeof(float));
    (void)ws_size;

    convert_kernel<<<64, 256, 0, stream>>>(W2, w_a, b_a, w_b, ps, ne,
                                           W2h, WAh, WBh, biasA);

    enc1_kernel<<<(N + 7) / 8, 256, 0, stream>>>(features, neigh1, W1, h1, N);

    int tiles = (N + 15) / 16;
    enc2_mlp_kernel<<<(tiles + 3) / 4, 128, 0, stream>>>(
        nodes, h1, neigh2, W2h, WAh, WBh, biasA, b_b, w_c, b_c, scores, N);

    softmax_kernel<<<1, 1024, 0, stream>>>(scores, (float*)d_out, N);
}
